// HDMODEL_618475291036
// MI455X (gfx1250) — matrix operand order repaired and checked by reading, not compiled, not run on hardware
//
#include <hip/hip_runtime.h>
#include <hip/hip_bf16.h>
#include <math.h>

// ---------------- problem constants ----------------
#define BB 16
#define MM 4096
#define DD 512
#define NN 8
#define PP 4
#define HH 1024
#define NP 32           // N*P
#define RTOT (BB*MM)    // 65536 token rows

typedef __attribute__((ext_vector_type(16))) __bf16 v16bf;
typedef __attribute__((ext_vector_type(8)))  float  v8f;

#define DI __device__ __forceinline__

DI __bf16 f2bf(float v) { return (__bf16)v; }

DI v8f v8zero() { v8f z = {0.f,0.f,0.f,0.f,0.f,0.f,0.f,0.f}; return z; }

DI v8f wmma_bf16(v16bf a, v16bf b, v8f c) {
  // D = A(16x32 bf16) x B(32x16 bf16) + C(16x16 f32)
  return __builtin_amdgcn_wmma_f32_16x16x32_bf16(false, a, false, b, (short)0, c, false, false);
}

// 32-byte fragment loaders (lower to ds_load_b128 / global_load_b128 pairs)
struct Frag32 { uint4 lo, hi; };

DI v16bf ld_b_contig(const __bf16* p) {        // B frag: 8 at p, 8 at p+16
  Frag32 f;
  f.lo = *(const uint4*)p;
  f.hi = *(const uint4*)(p + 16);
  return __builtin_bit_cast(v16bf, f);
}

DI v16bf ld_a_split(const __bf16* p) {         // A frag: 8 at p, 8 at p+16
  Frag32 f;
  f.lo = *(const uint4*)p;
  f.hi = *(const uint4*)(p + 16);
  return __builtin_bit_cast(v16bf, f);
}

// XOR block swizzle (8-element granularity) to avoid LDS bank conflicts
// without padding: row r, element k -> swizzled element index.
DI int swzk(int row, int k) {
  return ((((k >> 3) ^ (row & 7)) & 63) << 3) | (k & 7);
}

// =====================================================================
// K1: per-token inverse L2 norm of x rows  (one wave32 per row)
// =====================================================================
__global__ __launch_bounds__(256) void k_rownorm(const float* __restrict__ x,
                                                 float* __restrict__ invn) {
  int row  = blockIdx.x * 8 + (threadIdx.x >> 5);
  int lane = threadIdx.x & 31;
  const float* xr = x + (size_t)row * DD;
  float s = 0.f;
  #pragma unroll 4
  for (int i = lane; i < DD; i += 32) { float v = xr[i]; s += v * v; }
  #pragma unroll
  for (int off = 16; off > 0; off >>= 1) s += __shfl_xor(s, off, 32);
  if (lane == 0) invn[row] = 1.0f / fmaxf(sqrtf(s), 1e-12f);
}

// =====================================================================
// K2: phin = scale * l2norm(phi, axis=0)  -> bf16 [D, 32]
// =====================================================================
__global__ __launch_bounds__(256) void k_phinorm(const float* __restrict__ phi,
                                                 const float* __restrict__ scale,
                                                 __bf16* __restrict__ phin) {
  __shared__ float inv[NP];
  if (threadIdx.x < NP) {
    float s = 0.f;
    for (int d = 0; d < DD; ++d) { float v = phi[d * NP + threadIdx.x]; s += v * v; }
    inv[threadIdx.x] = 1.0f / fmaxf(sqrtf(s), 1e-12f);
  }
  __syncthreads();
  float sc = scale[0];
  for (int i = threadIdx.x; i < DD * NP; i += 256)
    phin[i] = f2bf(phi[i] * sc * inv[i & 31]);
}

// =====================================================================
// K3: logits[b,m,np] = xn @ phin   (65536x512 * 512x32, bf16 WMMA)
// block = 128 thr = 4 waves, 64 rows/block
// phin staged TRANSPOSED in LDS (col-major, padded rows) so B fragments
// are contiguous 32-byte ds_load_b128 pairs.
// =====================================================================
#define PSTRIDE 520   // DD + 8 bf16 pad: 260 words/row -> conflict-free
__global__ __launch_bounds__(128) void k_logits(const float* __restrict__ x,
                                                const float* __restrict__ invn,
                                                const __bf16* __restrict__ phin,
                                                float* __restrict__ logits) {
  __shared__ __bf16 sBt[NP * PSTRIDE];           // ~32.5 KB, [col][k]
  for (int i = threadIdx.x; i < DD * NP; i += 128) {
    int k = i >> 5, np = i & 31;
    sBt[np * PSTRIDE + k] = phin[i];
  }
  __syncthreads();

  const int wave = threadIdx.x >> 5;
  const int lane = threadIdx.x & 31;
  const int rowbase = blockIdx.x * 64 + wave * 16;
  const int ar   = lane & 15;            // A row within tile
  const int koff = (lane >> 4) << 3;     // 0 / 8  (A K sub-offset)
  const int bc   = lane & 15;            // B column within tile
  const int bko  = (lane >> 4) << 3;     // 0 / 8  (B K sub-offset)

  const int rowg = rowbase + ar;
  const float inv = invn[rowg];
  const float* xrow = x + (size_t)rowg * DD;
  const __bf16* brow0 = sBt + (size_t)bc * PSTRIDE + bko;
  const __bf16* brow1 = sBt + (size_t)(16 + bc) * PSTRIDE + bko;

  v8f acc0 = v8zero(), acc1 = v8zero();

  for (int kk = 0; kk < DD; kk += 32) {
    // ---- A fragment: xn tile 16x32, built on the fly ----
    v16bf a;
    const float4* p0 = (const float4*)(xrow + kk + koff);
    const float4* p1 = (const float4*)(xrow + kk + koff + 16);
    float4 f0 = p0[0], f1 = p0[1], f2 = p1[0], f3 = p1[1];
    a[0]=f2bf(f0.x*inv); a[1]=f2bf(f0.y*inv); a[2]=f2bf(f0.z*inv); a[3]=f2bf(f0.w*inv);
    a[4]=f2bf(f1.x*inv); a[5]=f2bf(f1.y*inv); a[6]=f2bf(f1.z*inv); a[7]=f2bf(f1.w*inv);
    a[8]=f2bf(f2.x*inv); a[9]=f2bf(f2.y*inv); a[10]=f2bf(f2.z*inv); a[11]=f2bf(f2.w*inv);
    a[12]=f2bf(f3.x*inv); a[13]=f2bf(f3.y*inv); a[14]=f2bf(f3.z*inv); a[15]=f2bf(f3.w*inv);
    // ---- B fragments: contiguous 32B LDS loads ----
    v16bf b0  = ld_b_contig(brow0 + kk);
    v16bf b1v = ld_b_contig(brow1 + kk);
    acc0 = wmma_bf16(a, b0,  acc0);
    acc1 = wmma_bf16(a, b1v, acc1);
  }
  // ---- store C (f32 16x16 layout: VGPR r -> M=r / M=8+r) ----
  const int crow = rowbase + ((lane >> 4) << 3);
  const int ccol = lane & 15;
  #pragma unroll
  for (int r = 0; r < 8; ++r) {
    logits[(size_t)(crow + r) * NP + ccol]      = acc0[r];
    logits[(size_t)(crow + r) * NP + 16 + ccol] = acc1[r];
  }
}

// =====================================================================
// K4: per-(b,np) online softmax stats over m; per-b flat-softmax stats
// =====================================================================
__global__ __launch_bounds__(1024) void k_colstats(const float* __restrict__ logits,
                                                   float* __restrict__ colmax,
                                                   float* __restrict__ invcolsum,
                                                   float* __restrict__ gstats,
                                                   float* __restrict__ p_t,
                                                   float* __restrict__ mi_out) {
  int b   = blockIdx.x;
  int col = threadIdx.x & 31;
  int sub = threadIdx.x >> 5;                    // 0..31
  const float* lg = logits + (size_t)b * MM * NP;
  float m = -1e30f, s = 0.f;
  for (int mr = sub; mr < MM; mr += 32) {
    float v = lg[mr * NP + col];
    if (v > m) { s = s * __expf(m - v) + 1.f; m = v; }
    else       { s += __expf(v - m); }
  }
  __shared__ float sm[1024], ss[1024];
  sm[threadIdx.x] = m; ss[threadIdx.x] = s;
  __syncthreads();
  for (int stride = 16; stride > 0; stride >>= 1) {
    if (sub < stride) {
      int i0 = sub * 32 + col, i1 = (sub + stride) * 32 + col;
      float m0 = sm[i0], s0 = ss[i0], m1 = sm[i1], s1 = ss[i1];
      float Mx = fmaxf(m0, m1);
      sm[i0] = Mx; ss[i0] = s0 * __expf(m0 - Mx) + s1 * __expf(m1 - Mx);
    }
    __syncthreads();
  }
  if (threadIdx.x < 32) {
    colmax[b * NP + col]    = sm[col];
    invcolsum[b * NP + col] = 1.0f / ss[col];
  }
  __syncthreads();
  if (threadIdx.x == 0) {
    float gm = -1e30f;
    for (int c2 = 0; c2 < 32; ++c2) gm = fmaxf(gm, sm[c2]);
    float gz = 0.f;
    for (int c2 = 0; c2 < 32; ++c2) gz += ss[c2] * __expf(sm[c2] - gm);
    gstats[b * 2] = gm; gstats[b * 2 + 1] = gz;
  }
  if (threadIdx.x < PP) p_t[b * PP + threadIdx.x] = 0.f;
  if (b == 0 && threadIdx.x == 0) *mi_out = 0.f;
}

// =====================================================================
// K5: per-row softmax -> c output; flat-softmax marginals p_mt, p_t
// =====================================================================
__global__ __launch_bounds__(256) void k_softmax_rows(const float* __restrict__ logits,
                                                      const float* __restrict__ gstats,
                                                      float* __restrict__ c_out,
                                                      float* __restrict__ p_mt,
                                                      float* __restrict__ p_t) {
  __shared__ float spt[PP];
  if (threadIdx.x < PP) spt[threadIdx.x] = 0.f;
  __syncthreads();
  int row = blockIdx.x * 256 + threadIdx.x;      // (b,m)
  int b = row >> 12;
  float v[NP];
  const float4* lp = (const float4*)(logits + (size_t)row * NP);
  #pragma unroll
  for (int i = 0; i < 8; ++i) {
    float4 f = lp[i];
    v[4*i] = f.x; v[4*i+1] = f.y; v[4*i+2] = f.z; v[4*i+3] = f.w;
  }
  float rmax = -1e30f;
  #pragma unroll
  for (int i = 0; i < NP; ++i) rmax = fmaxf(rmax, v[i]);
  float ex[NP]; float rsum = 0.f;
  #pragma unroll
  for (int i = 0; i < NP; ++i) { ex[i] = __expf(v[i] - rmax); rsum += ex[i]; }
  float rinv = 1.0f / rsum;
  float4* crow = (float4*)(c_out + (size_t)row * NP);
  #pragma unroll
  for (int i = 0; i < 8; ++i) {
    float4 cv; cv.x = ex[4*i]*rinv; cv.y = ex[4*i+1]*rinv;
    cv.z = ex[4*i+2]*rinv; cv.w = ex[4*i+3]*rinv;
    crow[i] = cv;
  }
  float gm = gstats[b * 2];
  float invgz = 1.0f / gstats[b * 2 + 1];
  float pm4[PP] = {0.f, 0.f, 0.f, 0.f};
  #pragma unroll
  for (int n = 0; n < NN; ++n)
    #pragma unroll
    for (int p = 0; p < PP; ++p)
      pm4[p] += __expf(v[n * PP + p] - gm);
  #pragma unroll
  for (int p = 0; p < PP; ++p) {
    pm4[p] *= invgz;
    p_mt[(size_t)row * PP + p] = pm4[p];
    atomicAdd(&spt[p], pm4[p]);
  }
  __syncthreads();
  if (threadIdx.x < PP) atomicAdd(&p_t[b * PP + threadIdx.x], spt[threadIdx.x]);
}

// =====================================================================
// K6: MI loss reduction
// =====================================================================
__global__ __launch_bounds__(256) void k_mi(const float* __restrict__ p_mt,
                                            const float* __restrict__ p_t,
                                            float* __restrict__ mi_out) {
  int row = blockIdx.x * 256 + threadIdx.x;
  int b = row >> 12;
  const float* pm = p_mt + (size_t)row * PP;
  float a0 = pm[0], a1 = pm[1], a2 = pm[2], a3 = pm[3];
  float p_m = a0 + a1 + a2 + a3;
  float acc = 0.f;
  acc += a0 * __logf(a0 / (p_m * p_t[b * PP + 0]) + 1e-10f);
  acc += a1 * __logf(a1 / (p_m * p_t[b * PP + 1]) + 1e-10f);
  acc += a2 * __logf(a2 / (p_m * p_t[b * PP + 2]) + 1e-10f);
  acc += a3 * __logf(a3 / (p_m * p_t[b * PP + 3]) + 1e-10f);
  __shared__ float red[256];
  red[threadIdx.x] = acc;
  __syncthreads();
  for (int s = 128; s > 0; s >>= 1) {
    if (threadIdx.x < s) red[threadIdx.x] += red[threadIdx.x + s];
    __syncthreads();
  }
  if (threadIdx.x == 0) atomicAdd(mi_out, -red[0]);
}

// =====================================================================
// K7: xs = d^T @ xn per batch (+task_emb).  Both LDS tiles stored
// transposed/padded so every fragment is a contiguous b128 pair.
// =====================================================================
#define DSTR 40   // 32 + 8 pad (20 words/row -> conflict-free)
__global__ __launch_bounds__(256) void k_xs(const float* __restrict__ x,
                                            const float* __restrict__ invn,
                                            const float* __restrict__ logits,
                                            const float* __restrict__ colmax,
                                            const float* __restrict__ invcolsum,
                                            const float* __restrict__ task_emb,
                                            float* __restrict__ xs) {
  int b = blockIdx.x >> 2;
  int dchunk = blockIdx.x & 3;                   // 128 cols each
  __shared__ __bf16 sDt[NP * DSTR];              // d^T tile  [np][m], 2.5 KB
  __shared__ __bf16 sXt[128 * DSTR];             // xn^T tile [d][m], 10 KB
  int wave = threadIdx.x >> 5, lane = threadIdx.x & 31;
  int ar = lane & 15, koff = (lane >> 4) << 3;
  int bc = lane & 15, bko = (lane >> 4) << 3;
  v8f acc0 = v8zero(), acc1 = v8zero();
  const float* lg = logits + (size_t)b * MM * NP;
  const float* cm = colmax + b * NP;
  const float* ic = invcolsum + b * NP;

  for (int kb = 0; kb < MM; kb += 32) {
    { // stage dispatch weights d = exp(logit-colmax)*invcolsum, transposed
      int idx = threadIdx.x * 4;
      int ml = idx >> 5, np0 = idx & 31;
      float4 lv = *(const float4*)(lg + (size_t)(kb + ml) * NP + np0);
      sDt[(np0 + 0) * DSTR + ml] = f2bf(__expf(lv.x - cm[np0 + 0]) * ic[np0 + 0]);
      sDt[(np0 + 1) * DSTR + ml] = f2bf(__expf(lv.y - cm[np0 + 1]) * ic[np0 + 1]);
      sDt[(np0 + 2) * DSTR + ml] = f2bf(__expf(lv.z - cm[np0 + 2]) * ic[np0 + 2]);
      sDt[(np0 + 3) * DSTR + ml] = f2bf(__expf(lv.w - cm[np0 + 3]) * ic[np0 + 3]);
    }
    { // stage normalized x tile, transposed
      int ml = threadIdx.x >> 3;
      int d0 = (threadIdx.x & 7) * 16;
      const float* xr = x + ((size_t)b * MM + kb + ml) * DD + dchunk * 128 + d0;
      float inv = invn[b * MM + kb + ml];
      #pragma unroll
      for (int j = 0; j < 4; ++j) {
        float4 f = ((const float4*)xr)[j];
        sXt[(d0 + 4*j + 0) * DSTR + ml] = f2bf(f.x * inv);
        sXt[(d0 + 4*j + 1) * DSTR + ml] = f2bf(f.y * inv);
        sXt[(d0 + 4*j + 2) * DSTR + ml] = f2bf(f.z * inv);
        sXt[(d0 + 4*j + 3) * DSTR + ml] = f2bf(f.w * inv);
      }
    }
    __syncthreads();
    // A = d^T (rows np): split 8+8 contiguous;  B = xn (k=m contiguous)
    v16bf a0 = ld_a_split(sDt + (size_t)ar * DSTR + koff);
    v16bf a1 = ld_a_split(sDt + (size_t)(16 + ar) * DSTR + koff);
    v16bf bB = ld_b_contig(sXt + (size_t)(wave * 16 + bc) * DSTR + bko);
    acc0 = wmma_bf16(a0, bB, acc0);
    acc1 = wmma_bf16(a1, bB, acc1);
    __syncthreads();
  }
  int ccol = dchunk * 128 + wave * 16 + (lane & 15);
  int rbase = (lane >> 4) << 3;
  #pragma unroll
  for (int r = 0; r < 8; ++r) {
    int np = rbase + r;
    int n = np >> 2, p = np & 3;
    xs[(((size_t)b * NN + n) * PP + p) * DD + ccol] = acc0[r] + task_emb[p * DD + ccol];
    int np2 = 16 + rbase + r;
    int n2 = np2 >> 2, p2 = np2 & 3;
    xs[(((size_t)b * NN + n2) * PP + p2) * DD + ccol] = acc1[r] + task_emb[p2 * DD + ccol];
  }
}

// =====================================================================
// K8: transpose+convert weights: W [NN][K][C] f32 -> Wt [NN][C][K] bf16
// (makes MLP B-fragments contiguous 32-byte global_load_b128 pairs)
// =====================================================================
__global__ __launch_bounds__(256) void k_wT(const float* __restrict__ W,
                                            __bf16* __restrict__ Wt,
                                            int K, int C) {
  size_t idx4 = ((size_t)blockIdx.x * 256 + threadIdx.x) * 4;
  size_t per = (size_t)K * C;
  int n = (int)(idx4 / per);
  size_t rem = idx4 % per;
  int k = (int)(rem / C);
  int c = (int)(rem % C);
  float4 f = *(const float4*)(W + idx4);
  __bf16* wt = Wt + (size_t)n * per;
  wt[(size_t)(c + 0) * K + k] = f2bf(f.x);
  wt[(size_t)(c + 1) * K + k] = f2bf(f.y);
  wt[(size_t)(c + 2) * K + k] = f2bf(f.z);
  wt[(size_t)(c + 3) * K + k] = f2bf(f.w);
}

// =====================================================================
// K9: per-expert GEMM1 + bias + silu -> Hmat bf16  ([64x512]*[512x1024])
// X_n staged in exactly 64 KB LDS with XOR bank swizzle; W1t col-major.
// =====================================================================
__global__ __launch_bounds__(256) void k_mlp1(const float* __restrict__ xs,
                                              const __bf16* __restrict__ W1t,
                                              const float* __restrict__ b1,
                                              __bf16* __restrict__ Hmat) {
  int n = blockIdx.x >> 2;
  int hchunk = blockIdx.x & 3;
  __shared__ __bf16 sX[64 * DD];                 // 64 KB, XOR-swizzled rows
  for (int idx = threadIdx.x; idx < 64 * DD / 4; idx += 256) {
    int row = idx >> 7;                          // 128 float4 per row
    int d4 = (idx & 127) * 4;
    int bb = row >> 2, p = row & 3;
    float4 f = *(const float4*)(xs + (((size_t)bb * NN + n) * PP + p) * DD + d4);
    __bf16* dst = sX + (size_t)row * DD + swzk(row, d4);   // 4 elems same block
    dst[0]=f2bf(f.x); dst[1]=f2bf(f.y); dst[2]=f2bf(f.z); dst[3]=f2bf(f.w);
  }
  __syncthreads();
  int wave = threadIdx.x >> 5, lane = threadIdx.x & 31;
  int ar = lane & 15, koff = (lane >> 4) << 3;
  int bc = lane & 15, bko = (lane >> 4) << 3;
  int c0 = hchunk * 256 + wave * 32;
  v8f acc[4][2];
  #pragma unroll
  for (int i = 0; i < 4; ++i) { acc[i][0] = v8zero(); acc[i][1] = v8zero(); }
  const __bf16* Wn = W1t + (size_t)n * HH * DD;
  for (int kk = 0; kk < DD; kk += 32) {
    v16bf a[4];
    #pragma unroll
    for (int rt = 0; rt < 4; ++rt) {
      int row = rt * 16 + ar;
      const __bf16* rb = sX + (size_t)row * DD;
      Frag32 fr;
      fr.lo = *(const uint4*)(rb + swzk(row, kk + koff));
      fr.hi = *(const uint4*)(rb + swzk(row, kk + 16 + koff));
      a[rt] = __builtin_bit_cast(v16bf, fr);
    }
    v16bf bb0 = ld_b_contig(Wn + (size_t)(c0 + bc) * DD + kk + bko);
    v16bf bb1 = ld_b_contig(Wn + (size_t)(c0 + 16 + bc) * DD + kk + bko);
    #pragma unroll
    for (int rt = 0; rt < 4; ++rt) {
      acc[rt][0] = wmma_bf16(a[rt], bb0, acc[rt][0]);
      acc[rt][1] = wmma_bf16(a[rt], bb1, acc[rt][1]);
    }
  }
  int rbase = (lane >> 4) << 3;
  #pragma unroll
  for (int rt = 0; rt < 4; ++rt)
    #pragma unroll
    for (int ct = 0; ct < 2; ++ct)
      #pragma unroll
      for (int r = 0; r < 8; ++r) {
        int row = rt * 16 + rbase + r;
        int col = c0 + ct * 16 + (lane & 15);
        float h = acc[rt][ct][r] + b1[n * HH + col];
        float sg = 1.0f / (1.0f + __expf(-h));
        Hmat[(size_t)n * 64 * HH + (size_t)row * HH + col] = f2bf(h * sg);
      }
}

// =====================================================================
// K10: per-expert GEMM2 + bias + residual -> rbuf ([64x1024]*[1024x512])
// A from Hmat (row-major: contiguous b128 pairs), B from W2t col-major.
// =====================================================================
__global__ __launch_bounds__(256) void k_mlp2(const __bf16* __restrict__ Hmat,
                                              const __bf16* __restrict__ W2t,
                                              const float* __restrict__ b2,
                                              const float* __restrict__ xs,
                                              float* __restrict__ rbuf) {
  int n = blockIdx.x >> 1;
  int dchunk = blockIdx.x & 1;
  int wave = threadIdx.x >> 5, lane = threadIdx.x & 31;
  int ar = lane & 15, koff = (lane >> 4) << 3;
  int bc = lane & 15, bko = (lane >> 4) << 3;
  int c0 = dchunk * 256 + wave * 32;
  v8f acc[4][2];
  #pragma unroll
  for (int i = 0; i < 4; ++i) { acc[i][0] = v8zero(); acc[i][1] = v8zero(); }
  const __bf16* Hn = Hmat + (size_t)n * 64 * HH;
  const __bf16* Wn = W2t + (size_t)n * DD * HH;
  for (int kk = 0; kk < HH; kk += 32) {
    v16bf a[4];
    #pragma unroll
    for (int rt = 0; rt < 4; ++rt)
      a[rt] = ld_a_split(Hn + (size_t)(rt * 16 + ar) * HH + kk + koff);
    v16bf bb0 = ld_b_contig(Wn + (size_t)(c0 + bc) * HH + kk + bko);
    v16bf bb1 = ld_b_contig(Wn + (size_t)(c0 + 16 + bc) * HH + kk + bko);
    #pragma unroll
    for (int rt = 0; rt < 4; ++rt) {
      acc[rt][0] = wmma_bf16(a[rt], bb0, acc[rt][0]);
      acc[rt][1] = wmma_bf16(a[rt], bb1, acc[rt][1]);
    }
  }
  int rbase = (lane >> 4) << 3;
  #pragma unroll
  for (int rt = 0; rt < 4; ++rt)
    #pragma unroll
    for (int ct = 0; ct < 2; ++ct)
      #pragma unroll
      for (int r = 0; r < 8; ++r) {
        int row = rt * 16 + rbase + r;
        int col = c0 + ct * 16 + (lane & 15);
        int bb = row >> 2, p = row & 3;
        size_t oi = (((size_t)bb * NN + n) * PP + p) * DD + col;
        rbuf[oi] = acc[rt][ct][r] + b2[n * DD + col] + xs[oi];
      }
}

// =====================================================================
// K11: LayerNorm in place on rbuf (512 rows of 512)
// =====================================================================
__global__ __launch_bounds__(256) void k_ln(float* __restrict__ r,
                                            const float* __restrict__ g,
                                            const float* __restrict__ bta) {
  int row = blockIdx.x;                          // ((b*8+n)*4+p)
  int n = (row >> 2) & 7;
  float* rp = r + (size_t)row * DD;
  float2 v = ((const float2*)rp)[threadIdx.x];
  float s = v.x + v.y;
  float q = v.x * v.x + v.y * v.y;
  #pragma unroll
  for (int off = 16; off > 0; off >>= 1) {
    s += __shfl_xor(s, off, 32);
    q += __shfl_xor(q, off, 32);
  }
  __shared__ float rs_[8], rq_[8];
  int wv = threadIdx.x >> 5, ln2 = threadIdx.x & 31;
  if (ln2 == 0) { rs_[wv] = s; rq_[wv] = q; }
  __syncthreads();
  if (threadIdx.x == 0) {
    float S = 0.f, Q = 0.f;
    for (int i = 0; i < 8; ++i) { S += rs_[i]; Q += rq_[i]; }
    rs_[0] = S; rq_[0] = Q;
  }
  __syncthreads();
  float mu = rs_[0] * (1.0f / DD);
  float var = rq_[0] * (1.0f / DD) - mu * mu;
  float rstd = rsqrtf(var + 1e-5f);
  int d = threadIdx.x * 2;
  float2 o;
  o.x = (v.x - mu) * rstd * g[n * DD + d]     + bta[n * DD + d];
  o.y = (v.y - mu) * rstd * g[n * DD + d + 1] + bta[n * DD + d + 1];
  ((float2*)rp)[threadIdx.x] = o;
}

// =====================================================================
// K12: y[b,m,p,:] = sum_n c[b,m,n,p] * ys[b,n,p,:]  (streaming, 537 MB)
// ys slice staged via GLOBAL_LOAD_ASYNC_TO_LDS_B128 (ASYNCcnt path).
// =====================================================================
__global__ __launch_bounds__(256) void k_combine(const float* __restrict__ ys,
                                                 const float* __restrict__ c_all,
                                                 float* __restrict__ y) {
  int b = blockIdx.x >> 6;
  int mchunk = blockIdx.x & 63;                  // 64 m per block
  __shared__ float sY[NN * PP * DD];             // 64 KB
  const float* yb = ys + (size_t)b * NN * PP * DD;
  // async byte-copy global->LDS: 16 x b128 per thread, tracked on ASYNCcnt
  for (int i = threadIdx.x * 4; i < NN * PP * DD; i += 1024) {
    unsigned lds_addr = (unsigned)(unsigned long long)(uintptr_t)(sY + i);
    unsigned long long gaddr = (unsigned long long)(uintptr_t)(yb + i);
    asm volatile("global_load_async_to_lds_b128 %0, %1, off"
                 :: "v"(lds_addr), "v"(gaddr) : "memory");
  }
  asm volatile("s_wait_asynccnt 0" ::: "memory");
  __syncthreads();
  int p = threadIdx.x >> 6;                      // 0..3
  int d0 = (threadIdx.x & 63) * 8;               // 8 floats per thread
  for (int mm = 0; mm < 64; ++mm) {
    int m = mchunk * 64 + mm;
    const float* cr = c_all + ((size_t)b * MM + m) * NP;
    if (mm + 1 < 64) __builtin_prefetch(cr + NP, 0, 1);
    float4 o0 = {0.f,0.f,0.f,0.f}, o1 = {0.f,0.f,0.f,0.f};
    #pragma unroll
    for (int n = 0; n < NN; ++n) {
      float w = cr[n * PP + p];
      const float* yv = sY + (n * PP + p) * DD + d0;
      float4 s0 = *(const float4*)yv;
      float4 s1 = *(const float4*)(yv + 4);
      o0.x += w * s0.x; o0.y += w * s0.y; o0.z += w * s0.z; o0.w += w * s0.w;
      o1.x += w * s1.x; o1.y += w * s1.y; o1.z += w * s1.z; o1.w += w * s1.w;
    }
    float* op = y + ((((size_t)b * MM + m) * PP + p) * DD) + d0;
    ((float4*)op)[0] = o0;
    ((float4*)op)[1] = o1;
  }
}

// =====================================================================
// host launcher
// =====================================================================
extern "C" void kernel_launch(void* const* d_in, const int* in_sizes, int n_in,
                              void* d_out, int out_size, void* d_ws, size_t ws_size,
                              hipStream_t stream) {
  const float* x        = (const float*)d_in[0];
  const float* phi      = (const float*)d_in[1];
  const float* scale    = (const float*)d_in[2];
  const float* task_emb = (const float*)d_in[3];
  const float* W1       = (const float*)d_in[4];
  const float* b1       = (const float*)d_in[5];
  const float* W2       = (const float*)d_in[6];
  const float* b2       = (const float*)d_in[7];
  const float* ln_g     = (const float*)d_in[8];
  const float* ln_b     = (const float*)d_in[9];

  float* y_out  = (float*)d_out;
  float* c_out  = y_out + (size_t)BB * MM * PP * DD;
  float* mi_out = c_out + (size_t)BB * MM * NP;

  char* ws = (char*)d_ws;
  size_t o = 0;
  auto take = [&](size_t bytes) { char* p = ws + o; o = (o + bytes + 255) & ~(size_t)255; return p; };
  float*  invn      = (float*) take((size_t)RTOT * 4);
  __bf16* phin      = (__bf16*)take((size_t)DD * NP * 2);
  float*  logits    = (float*) take((size_t)RTOT * NP * 4);
  float*  colmax    = (float*) take((size_t)BB * NP * 4);
  float*  invcolsum = (float*) take((size_t)BB * NP * 4);
  float*  gstats    = (float*) take((size_t)BB * 2 * 4);
  float*  p_t       = (float*) take((size_t)BB * PP * 4);
  float*  p_mt      = (float*) take((size_t)RTOT * PP * 4);
  float*  xs        = (float*) take((size_t)BB * NN * PP * DD * 4);
  __bf16* W1t       = (__bf16*)take((size_t)NN * DD * HH * 2);
  __bf16* W2t       = (__bf16*)take((size_t)NN * HH * DD * 2);
  __bf16* Hmat      = (__bf16*)take((size_t)NN * 64 * HH * 2);
  float*  rbuf      = (float*) take((size_t)BB * NN * PP * DD * 4);
  (void)ws_size; (void)in_sizes; (void)n_in; (void)out_size;

  k_rownorm     <<<RTOT / 8, 256, 0, stream>>>(x, invn);
  k_phinorm     <<<1, 256, 0, stream>>>(phi, scale, phin);
  k_logits      <<<RTOT / 64, 128, 0, stream>>>(x, invn, phin, logits);
  k_colstats    <<<BB, 1024, 0, stream>>>(logits, colmax, invcolsum, gstats, p_t, mi_out);
  k_softmax_rows<<<RTOT / 256, 256, 0, stream>>>(logits, gstats, c_out, p_mt, p_t);
  k_mi          <<<RTOT / 256, 256, 0, stream>>>(p_mt, p_t, mi_out);
  k_xs          <<<BB * 4, 256, 0, stream>>>(x, invn, logits, colmax, invcolsum, task_emb, xs);
  k_wT          <<<(NN * DD * HH) / (256 * 4), 256, 0, stream>>>(W1, W1t, DD, HH);
  k_wT          <<<(NN * HH * DD) / (256 * 4), 256, 0, stream>>>(W2, W2t, HH, DD);
  k_mlp1        <<<NN * 4, 256, 0, stream>>>(xs, W1t, b1, Hmat);
  k_mlp2        <<<NN * 2, 256, 0, stream>>>(Hmat, W2t, b2, xs, rbuf);
  k_ln          <<<BB * NN * PP, 256, 0, stream>>>(rbuf, ln_g, ln_b);
  k_combine     <<<BB * 64, 256, 0, stream>>>(rbuf, c_out, y_out);
}
